// Net_24670292148892
// MI455X (gfx1250) — compile-verified, matrix operand order checked by reading, not run on hardware
//
#include <hip/hip_runtime.h>

#define DEVINL __device__ __forceinline__

typedef __attribute__((ext_vector_type(16))) _Float16 v16h;
typedef __attribute__((ext_vector_type(8)))  float    v8f;

#define WMMA(A,B,C) __builtin_amdgcn_wmma_f32_16x16x32_f16(false,(A),false,(B),(short)0,(C),false,false)

// problem constants
#define BATCH 4096
#define TE 168
#define TD 24
#define XW 8
#define HD 64
#define NG 256       // 4*H gate width
#define BT 16        // batch rows per workgroup
#define HSTR 72      // ushort stride of h rows (144 B, 16B-aligned rows)
#define GSTR 264     // float stride of gate rows (bank-spread padding)

union HFrag {
  unsigned int u[8];
  _Float16     h[16];
  v16h         v;
};
union HalfBits { _Float16 h; unsigned short u; };

DEVINL unsigned short f2h_bits(float f){ HalfBits x; x.h = (_Float16)f; return x.u; }
DEVINL float h2f(unsigned short b){ HalfBits x; x.u = b; return (float)x.h; }

DEVINL float sigm(float x){ return 1.f/(1.f + __expf(-x)); }
DEVINL float tanhfast(float x){ return 2.f/(1.f + __expf(-2.f*x)) - 1.f; }
DEVINL float softplusf(float x){ return x > 15.f ? x : __logf(1.f + __expf(x)); }

// Combined-K weight fetch: gates = [x | h] @ [Wih^T ; Whh^T]
// layer 0: K 0..7 = Wih0^T (X=8), K 8..31 = zero pad, K 32..95 = Whh0^T
DEVINL float wfetch0(const float* Wih, const float* Whh, int k, int n){
  float r = 0.f;
  if (k < 8)        r = Wih[n*8 + k];
  else if (k >= 32) r = Whh[n*64 + (k - 32)];
  return r;
}
// layers 1/2: K 0..63 = Wih^T, K 64..127 = Whh^T
DEVINL float wfetch12(const float* Wih, const float* Whh, int k, int n){
  return (k < 64) ? Wih[n*64 + k] : Whh[n*64 + (k - 64)];
}

// B fragment (32x16 K x N slice), lane l: col = n, kb = (l>>4)*8,
// VGPR j<4 holds K = 32*kblk+kb+2j,(+1); j>=4 holds K = 32*kblk+16+kb+2(j-4),(+1)
DEVINL v16h load_wfrag0(const float* Wih, const float* Whh, int kblk, int col, int lane){
  const int kb = (lane >> 4) * 8;
  HFrag fr;
  #pragma unroll
  for (int j = 0; j < 8; ++j){
    int k = 32*kblk + ((j < 4) ? (kb + 2*j) : (16 + kb + 2*(j-4)));
    fr.h[2*j]   = (_Float16)wfetch0(Wih, Whh, k,   col);
    fr.h[2*j+1] = (_Float16)wfetch0(Wih, Whh, k+1, col);
  }
  return fr.v;
}
DEVINL v16h load_wfrag12(const float* Wih, const float* Whh, int kblk, int col, int lane){
  const int kb = (lane >> 4) * 8;
  HFrag fr;
  #pragma unroll
  for (int j = 0; j < 8; ++j){
    int k = 32*kblk + ((j < 4) ? (kb + 2*j) : (16 + kb + 2*(j-4)));
    fr.h[2*j]   = (_Float16)wfetch12(Wih, Whh, k,   col);
    fr.h[2*j+1] = (_Float16)wfetch12(Wih, Whh, k+1, col);
  }
  return fr.v;
}

// A fragment (16x32 M x K slice) of hidden state from LDS, two ds_load_b128 per lane
DEVINL v16h load_hfrag(const unsigned short* hbase, int k0, int lane){
  const int row = lane & 15;
  const int kb  = (lane >> 4) * 8;
  HFrag fr;
  const uint4 lo = *(const uint4*)(hbase + row*HSTR + k0 + kb);
  const uint4 hi = *(const uint4*)(hbase + row*HSTR + k0 + 16 + kb);
  fr.u[0]=lo.x; fr.u[1]=lo.y; fr.u[2]=lo.z; fr.u[3]=lo.w;
  fr.u[4]=hi.x; fr.u[5]=hi.y; fr.u[6]=hi.z; fr.u[7]=hi.w;
  return fr.v;
}

// A fragment for layer-0 x slot: K 0..7 = x row, K 8..31 = zeros
DEVINL v16h load_xfrag(const unsigned short* xb, int lane){
  HFrag fr;
  if (lane < 16){
    const uint4 lo = *(const uint4*)(xb + lane*8);
    fr.u[0]=lo.x; fr.u[1]=lo.y; fr.u[2]=lo.z; fr.u[3]=lo.w;
  } else {
    fr.u[0]=0; fr.u[1]=0; fr.u[2]=0; fr.u[3]=0;
  }
  fr.u[4]=0; fr.u[5]=0; fr.u[6]=0; fr.u[7]=0;
  return fr.v;
}

// D layout: VGPR r, lanes 0-15 -> M=r, lanes 16-31 -> M=r+8; N = lane&15 (+colbase)
DEVINL void store_gates(float* gb, int col, int lane, v8f acc){
  const int rbase = (lane >> 4) * 8;
  #pragma unroll
  for (int r = 0; r < 8; ++r) gb[(rbase + r)*GSTR + col] = acc[r];
}

DEVINL void cell_x_wmma(const unsigned short* xb, const unsigned short* hs,
                        const v16h (&Bf)[2][3], float* gb, int wave, int lane){
  v8f a0 = {}; v8f a1 = {};
  v16h a = load_xfrag(xb, lane);
  a0 = WMMA(a, Bf[0][0], a0); a1 = WMMA(a, Bf[1][0], a1);
  a = load_hfrag(hs, 0, lane);
  a0 = WMMA(a, Bf[0][1], a0); a1 = WMMA(a, Bf[1][1], a1);
  a = load_hfrag(hs, 32, lane);
  a0 = WMMA(a, Bf[0][2], a0); a1 = WMMA(a, Bf[1][2], a1);
  const int col0 = 32*wave + (lane & 15);
  store_gates(gb, col0,      lane, a0);
  store_gates(gb, col0 + 16, lane, a1);
}

DEVINL void cell_h_wmma(const unsigned short* hx, const unsigned short* hs,
                        const v16h (&Bf)[2][4], float* gb, int wave, int lane){
  v8f a0 = {}; v8f a1 = {};
  v16h a;
  a = load_hfrag(hx, 0, lane);  a0 = WMMA(a, Bf[0][0], a0); a1 = WMMA(a, Bf[1][0], a1);
  a = load_hfrag(hx, 32, lane); a0 = WMMA(a, Bf[0][1], a0); a1 = WMMA(a, Bf[1][1], a1);
  a = load_hfrag(hs, 0, lane);  a0 = WMMA(a, Bf[0][2], a0); a1 = WMMA(a, Bf[1][2], a1);
  a = load_hfrag(hs, 32, lane); a0 = WMMA(a, Bf[0][3], a0); a1 = WMMA(a, Bf[1][3], a1);
  const int col0 = 32*wave + (lane & 15);
  store_gates(gb, col0,      lane, a0);
  store_gates(gb, col0 + 16, lane, a1);
}

// element-wise: thread owns column j for 4 rows; c stays in registers across all steps
DEVINL void cell_ew(const float* gb, const float* bias, float creg[4],
                    unsigned short* hout, int j, int r0){
  #pragma unroll
  for (int k = 0; k < 4; ++k){
    const int row = r0 + k;
    const float* gr = gb + row*GSTR;
    float iv = sigm    (gr[j]       + bias[j]);
    float fv = sigm    (gr[64 + j]  + bias[64 + j]);
    float gv = tanhfast(gr[128 + j] + bias[128 + j]);
    float ov = sigm    (gr[192 + j] + bias[192 + j]);
    float c  = fv*creg[k] + iv*gv;
    creg[k]  = c;
    hout[row*HSTR + j] = f2h_bits(ov * tanhfast(c));
  }
}

__global__ __launch_bounds__(256, 1)
void lstm_forecast_kernel(const float* __restrict__ enc_x, const float* __restrict__ enc_z,
                          const float* __restrict__ dec_x, const float* __restrict__ vbuf,
                          const float* __restrict__ eps,
                          const float* __restrict__ Wih0, const float* __restrict__ Whh0, const float* __restrict__ b0v,
                          const float* __restrict__ Wih1, const float* __restrict__ Whh1, const float* __restrict__ b1v,
                          const float* __restrict__ Wih2, const float* __restrict__ Whh2, const float* __restrict__ b2v,
                          const float* __restrict__ w_m, const float* __restrict__ b_m,
                          const float* __restrict__ w_a, const float* __restrict__ b_a,
                          float* __restrict__ out)
{
  __shared__ __align__(16) unsigned short s_h[3][BT][HSTR];  // f16 hidden states
  __shared__ __align__(16) unsigned short s_x[BT][8];        // f16 layer-0 input
  __shared__ float s_g[BT][GSTR];                            // f32 gate tile
  __shared__ float s_bias[3][NG];
  __shared__ float s_wm[HD], s_wa[HD];
  __shared__ float s_vf[BT], s_z[BT], s_dm[BT], s_da[BT];
  __shared__ float s_bm, s_ba;

  const int tid  = threadIdx.x;
  const int lane = tid & 31;
  const int wave = tid >> 5;
  const int b0   = blockIdx.x * BT;

  // one-time LDS staging
  for (int i = tid; i < 3*BT*HSTR; i += 256) ((unsigned short*)s_h)[i] = 0;
  s_bias[0][tid] = b0v[tid];
  s_bias[1][tid] = b1v[tid];
  s_bias[2][tid] = b2v[tid];
  if (tid < HD){ s_wm[tid] = w_m[tid]; s_wa[tid] = w_a[tid]; }
  if (tid < BT){ s_vf[tid] = vbuf[b0 + tid]; }
  if (tid == 0){ s_bm = b_m[0]; s_ba = b_a[0]; }

  // register-resident f16 weight fragments (reused 192x)
  const int col0 = 32*wave + (lane & 15);
  const int col1 = col0 + 16;
  v16h B0f[2][3], B1f[2][4], B2f[2][4];
  #pragma unroll
  for (int kb2 = 0; kb2 < 3; ++kb2){
    B0f[0][kb2] = load_wfrag0(Wih0, Whh0, kb2, col0, lane);
    B0f[1][kb2] = load_wfrag0(Wih0, Whh0, kb2, col1, lane);
  }
  #pragma unroll
  for (int kb2 = 0; kb2 < 4; ++kb2){
    B1f[0][kb2] = load_wfrag12(Wih1, Whh1, kb2, col0, lane);
    B1f[1][kb2] = load_wfrag12(Wih1, Whh1, kb2, col1, lane);
    B2f[0][kb2] = load_wfrag12(Wih2, Whh2, kb2, col0, lane);
    B2f[1][kb2] = load_wfrag12(Wih2, Whh2, kb2, col1, lane);
  }

  // per-thread register-resident cell state
  float c0r[4] = {0,0,0,0}, c1r[4] = {0,0,0,0}, c2r[4] = {0,0,0,0};
  const int ew_j  = tid & 63;
  const int ew_r0 = (tid >> 6) * 4;

  float* gb = &s_g[0][0];
  unsigned short* h0 = &s_h[0][0][0];
  unsigned short* h1 = &s_h[1][0][0];
  unsigned short* h2 = &s_h[2][0][0];
  unsigned short* xb = &s_x[0][0];

  __syncthreads();

  // ---------------- encoder ----------------
  for (int t = 0; t < TE; ++t){
    if (tid < 128){
      int row = tid >> 3, kk = tid & 7;
      xb[row*8 + kk] = f2h_bits(enc_x[(size_t)(b0+row)*(TE*XW) + (size_t)t*XW + kk]);
    }
    __syncthreads();
    cell_x_wmma(xb, h0, B0f, gb, wave, lane);
    __syncthreads();
    cell_ew(gb, s_bias[0], c0r, h0, ew_j, ew_r0);
    __syncthreads();
    cell_h_wmma(h0, h1, B1f, gb, wave, lane);
    __syncthreads();
    cell_ew(gb, s_bias[1], c1r, h1, ew_j, ew_r0);
    __syncthreads();
    cell_h_wmma(h1, h2, B2f, gb, wave, lane);
    __syncthreads();
    cell_ew(gb, s_bias[2], c2r, h2, ew_j, ew_r0);
    __syncthreads();
  }

  // z = enc_z[:, -1] / v
  if (tid < BT) s_z[tid] = enc_z[(size_t)(b0+tid)*TE + (TE-1)] / s_vf[tid];
  __syncthreads();

  // ---------------- decoder ----------------
  for (int td = 0; td < TD; ++td){
    if (tid < 128){
      int row = tid >> 3, kk = tid & 7;
      float xv = (kk == 0) ? s_z[row]
                           : dec_x[(size_t)(b0+row)*(TD*(XW-1)) + (size_t)td*(XW-1) + (kk-1)];
      xb[row*8 + kk] = f2h_bits(xv);
    }
    __syncthreads();
    cell_x_wmma(xb, h0, B0f, gb, wave, lane);
    __syncthreads();
    cell_ew(gb, s_bias[0], c0r, h0, ew_j, ew_r0);
    __syncthreads();
    cell_h_wmma(h0, h1, B1f, gb, wave, lane);
    __syncthreads();
    cell_ew(gb, s_bias[1], c1r, h1, ew_j, ew_r0);
    __syncthreads();
    cell_h_wmma(h1, h2, B2f, gb, wave, lane);
    __syncthreads();
    cell_ew(gb, s_bias[2], c2r, h2, ew_j, ew_r0);
    __syncthreads();
    // head: m = (h2@w_m + b_m)*v ; a = softplus(h2@w_a + b_a)*v ; zs = m + a*eps
    if (tid < 32){
      int row = tid & 15, which = tid >> 4;
      const float* wv = which ? s_wa : s_wm;
      float s = which ? s_ba : s_bm;
      #pragma unroll 8
      for (int j2 = 0; j2 < HD; ++j2) s += h2f(h2[row*HSTR + j2]) * wv[j2];
      if (which) s_da[row] = s; else s_dm[row] = s;
    }
    __syncthreads();
    if (tid < BT){
      float vf = s_vf[tid];
      float m  = s_dm[tid] * vf;
      float aa = softplusf(s_da[tid]) * vf;
      float e  = eps[(size_t)(b0+tid)*TD + td];
      float zs = m + aa*e;
      out[(size_t)(b0+tid)*TD + td] = zs;
      s_z[tid] = zs / vf;
    }
    __syncthreads();
  }
}

extern "C" void kernel_launch(void* const* d_in, const int* in_sizes, int n_in,
                              void* d_out, int out_size, void* d_ws, size_t ws_size,
                              hipStream_t stream) {
  (void)in_sizes; (void)n_in; (void)out_size; (void)d_ws; (void)ws_size;
  const float* enc_x = (const float*)d_in[0];
  const float* enc_z = (const float*)d_in[1];
  const float* dec_x = (const float*)d_in[2];
  const float* v     = (const float*)d_in[3];
  const float* eps   = (const float*)d_in[4];
  const float* Wih0  = (const float*)d_in[5];
  const float* Whh0  = (const float*)d_in[6];
  const float* b0v   = (const float*)d_in[7];
  const float* Wih1  = (const float*)d_in[8];
  const float* Whh1  = (const float*)d_in[9];
  const float* b1v   = (const float*)d_in[10];
  const float* Wih2  = (const float*)d_in[11];
  const float* Whh2  = (const float*)d_in[12];
  const float* b2v   = (const float*)d_in[13];
  const float* w_m   = (const float*)d_in[14];
  const float* b_m   = (const float*)d_in[15];
  const float* w_a   = (const float*)d_in[16];
  const float* b_a   = (const float*)d_in[17];
  float* out = (float*)d_out;

  dim3 grid(BATCH / BT);   // 256 persistent workgroups, one 16-row batch tile each
  hipLaunchKernelGGL(lstm_forecast_kernel, grid, dim3(256), 0, stream,
                     enc_x, enc_z, dec_x, v, eps,
                     Wih0, Whh0, b0v, Wih1, Whh1, b1v, Wih2, Whh2, b2v,
                     w_m, b_m, w_a, b_a, out);
}
